// SGAN_decoder_9740985827965
// MI455X (gfx1250) — matrix operand order repaired and checked by reading, not compiled, not run on hardware
//
#include <hip/hip_runtime.h>
#include <hip/hip_bf16.h>
#include <math.h>

// Problem constants (from reference)
#define BATCH 512
#define H 64
#define E 64
#define HP 64
#define OBS 8
#define PRE 12

typedef __attribute__((ext_vector_type(16))) _Float16 v16h;
typedef __attribute__((ext_vector_type(8)))  _Float16 v8h;
typedef __attribute__((ext_vector_type(8)))  float    v8f;
typedef __attribute__((ext_vector_type(4)))  float    v4f;

// ---------------------------------------------------------------------------
// Kernel 0: one-time init.
//  - zero h (GRU carry starts at 0; spool step0 uses hidden_state separately)
//  - copy group_track into out[:, 0:8, :]
//  - last = group_track[:, 7, :]
//  - convert the FULL W_pool (128x64) to f16: rows 0..63 multiply E, rows
//    64..127 multiply h_j (both folded into one K=128 WMMA chain)
// ---------------------------------------------------------------------------
__global__ void k_init(const float* __restrict__ gt,
                       const float* __restrict__ W_pool,
                       float* __restrict__ h, float* __restrict__ last,
                       _Float16* __restrict__ wp16, float* __restrict__ out) {
    int t = blockIdx.x * blockDim.x + threadIdx.x;          // 32768 threads
    if (t < BATCH * H) h[t] = 0.0f;
    if (t < BATCH * OBS * 2) {                              // out row = 20 pairs
        int i = t >> 4, off = t & 15;
        out[i * 40 + off] = gt[t];
    }
    if (t < (HP + H) * H) wp16[t] = (_Float16)W_pool[t];    // all 128 rows
    if (t < BATCH * 2) {
        int i = t >> 1;
        last[t] = gt[i * 16 + 14 + (t & 1)];                // obs index 7
    }
}

// ---------------------------------------------------------------------------
// Kernel 1 (per step): prep
//   u[j][k]  = last[j] @ W_rel            (B x 64, f32)
//   h16[j][k] = f16(h_spool[j][k])        (B x 64, f16 A-matrix feed)
// ---------------------------------------------------------------------------
__global__ void k_prep(const float* __restrict__ last,
                       const float* __restrict__ hsp,
                       const float* __restrict__ W_rel,
                       float* __restrict__ u, _Float16* __restrict__ h16) {
    int t = blockIdx.x * blockDim.x + threadIdx.x;          // B*64 = 32768
    int j = t >> 6, n = t & 63;
    u[t] = last[j * 2] * W_rel[n] + last[j * 2 + 1] * W_rel[64 + n];
    h16[t] = (_Float16)hsp[t];
}

// ---------------------------------------------------------------------------
// Kernel 2 (per step): WMMA social pooling. One block per i, 8 waves.
//   K=128 GEMM: A = [ relu(u[j]-u[i]+b_rel) | h16[j] ]  (16x128 per j-tile)
//   4 x v_wmma_f32_16x16x32_f16 per n-tile (16 per j-tile), running col-max,
//   pooled[i][n] = relu(max_j(acc) + b_pool[n])
// ---------------------------------------------------------------------------
__global__ void __launch_bounds__(256)
k_spool(const float* __restrict__ u, const _Float16* __restrict__ h16,
        const float* __restrict__ b_rel, const float* __restrict__ b_pool,
        const _Float16* __restrict__ wp16, float* __restrict__ pooled) {
    __shared__ float cib[64];        // b_rel[k] - u[i][k]
    __shared__ float sbp[64];        // b_pool
    __shared__ float smax[8][64];    // per-wave column maxima

    const int i    = blockIdx.x;
    const int tid  = threadIdx.x;
    const int lane = tid & 31;
    const int w    = tid >> 5;       // wave id (wave32)
    const int m    = lane & 15;      // row/col within tile
    const int hi   = lane >> 4;      // lane half (0/1)

    if (tid < 64) {
        cib[tid] = b_rel[tid] - u[i * 64 + tid];
        sbp[tid] = b_pool[tid];
    }
    __syncthreads();

    // Resident B tiles Bt[nt*4+kc]: 32x16 f16 each, covering K=0..127.
    // ISA B layout: lane half 'hi' holds K = 8*hi + hh (hh<8), 16 + 8*hi + (hh-8) (hh>=8), N = m.
    v16h Bt[16];
#pragma unroll
    for (int nt = 0; nt < 4; ++nt)
#pragma unroll
        for (int kc = 0; kc < 4; ++kc)
#pragma unroll
            for (int hh = 0; hh < 16; ++hh) {
                int k = kc * 32 + (hh < 8 ? hi * 8 + hh : 16 + hi * 8 + (hh - 8));
                Bt[nt * 4 + kc][hh] = wp16[k * 64 + nt * 16 + m];
            }

    v8f runmax[4];
#pragma unroll
    for (int nt = 0; nt < 4; ++nt)
#pragma unroll
        for (int r = 0; r < 8; ++r)
            runmax[nt][r] = -3.0e38f;

    // 32 j-tiles split over 8 waves
    for (int jt = w; jt < 32; jt += 8) {
        const int j = jt * 16 + m;               // A row for this lane
        const float*    urow = u   + j * 64;
        const _Float16* hrow = h16 + j * 64;

        // A tiles. ISA 16-bit A 16x32 layout: within a 32-chunk,
        // halves 0..7  -> K = hi*8 + hh        (contiguous run of 8)
        // halves 8..15 -> K = 16 + hi*8 + (hh-8) (contiguous run of 8)
        v16h A[4];
#pragma unroll
        for (int kc = 0; kc < 2; ++kc) {        // E = relu(u[j]+cib), f32->f16
            const int k0 = kc * 32 + hi * 8;
            v4f ua = *(const v4f*)(urow + k0);
            v4f ub = *(const v4f*)(urow + k0 + 4);
            v4f uc = *(const v4f*)(urow + k0 + 16);
            v4f ud = *(const v4f*)(urow + k0 + 20);
            v4f ca = *(const v4f*)(cib + k0);
            v4f cb = *(const v4f*)(cib + k0 + 4);
            v4f cc = *(const v4f*)(cib + k0 + 16);
            v4f cd = *(const v4f*)(cib + k0 + 20);
            v4f e0 = ua + ca, e1 = ub + cb, e2 = uc + cc, e3 = ud + cd;
#pragma unroll
            for (int x = 0; x < 4; ++x) {
                A[kc][x]      = (_Float16)fmaxf(e0[x], 0.0f);
                A[kc][4 + x]  = (_Float16)fmaxf(e1[x], 0.0f);
                A[kc][8 + x]  = (_Float16)fmaxf(e2[x], 0.0f);
                A[kc][12 + x] = (_Float16)fmaxf(e3[x], 0.0f);
            }
        }
#pragma unroll
        for (int kc = 0; kc < 2; ++kc) {        // h_j half: straight f16 loads
            const int k0 = kc * 32 + hi * 8;
            v8h lo = *(const v8h*)(hrow + k0);
            v8h hi8 = *(const v8h*)(hrow + k0 + 16);
            A[2 + kc] = __builtin_shufflevector(lo, hi8,
                0, 1, 2, 3, 4, 5, 6, 7, 8, 9, 10, 11, 12, 13, 14, 15);
        }

#pragma unroll
        for (int nt = 0; nt < 4; ++nt) {
            v8f acc = {};
            acc = __builtin_amdgcn_wmma_f32_16x16x32_f16(
                false, A[0], false, Bt[nt * 4 + 0], (short)0, acc, false, false);
            acc = __builtin_amdgcn_wmma_f32_16x16x32_f16(
                false, A[1], false, Bt[nt * 4 + 1], (short)0, acc, false, false);
            acc = __builtin_amdgcn_wmma_f32_16x16x32_f16(
                false, A[2], false, Bt[nt * 4 + 2], (short)0, acc, false, false);
            acc = __builtin_amdgcn_wmma_f32_16x16x32_f16(
                false, A[3], false, Bt[nt * 4 + 3], (short)0, acc, false, false);
#pragma unroll
            for (int r = 0; r < 8; ++r)
                runmax[nt][r] = fmaxf(runmax[nt][r], acc[r]);
        }
    }

    // Reduce: 8 accum rows -> 1, lane-half merge, + b_pool, relu.
    float val[4];
#pragma unroll
    for (int nt = 0; nt < 4; ++nt) {
        float mx = runmax[nt][0];
#pragma unroll
        for (int r = 1; r < 8; ++r) mx = fmaxf(mx, runmax[nt][r]);
        mx = fmaxf(mx, __shfl_xor(mx, 16, 32));
        val[nt] = fmaxf(mx + sbp[nt * 16 + m], 0.0f);
    }
    if (lane < 16)
#pragma unroll
        for (int nt = 0; nt < 4; ++nt) smax[w][nt * 16 + lane] = val[nt];
    __syncthreads();

    if (tid < 64) {
        float mx = smax[0][tid];
#pragma unroll
        for (int ww = 1; ww < 8; ++ww) mx = fmaxf(mx, smax[ww][tid]);
        pooled[i * 64 + tid] = mx;
    }
}

// ---------------------------------------------------------------------------
// Kernel 3 (per step): head. One block per i, 64 threads.
//   pos = pooled @ W_pos + b_pos; emb = pos @ W_emb + b_emb; GRU step.
//   Writes pos into out[:, 8+step, :], pos -> last, h_new -> h.
// ---------------------------------------------------------------------------
__device__ __forceinline__ float sigmoidf_(float x) {
    return 1.0f / (1.0f + __expf(-x));
}

__global__ void k_head(const float* __restrict__ pooled,
                       const float* __restrict__ W_pos, const float* __restrict__ b_pos,
                       const float* __restrict__ W_emb, const float* __restrict__ b_emb,
                       const float* __restrict__ W_ih,  const float* __restrict__ W_hh,
                       const float* __restrict__ b_ih,  const float* __restrict__ b_hh,
                       float* __restrict__ h, float* __restrict__ last,
                       float* __restrict__ out, int step) {
    __shared__ float sp[64], semb[64], sh[64], spos[2];
    const int i = blockIdx.x, t = threadIdx.x;
    sp[t] = pooled[i * 64 + t];
    sh[t] = h[i * 64 + t];
    __syncthreads();
    if (t < 2) {
        float a = b_pos[t];
#pragma unroll 8
        for (int k = 0; k < 64; ++k) a += sp[k] * W_pos[k * 2 + t];
        spos[t] = a;
    }
    __syncthreads();
    semb[t] = spos[0] * W_emb[t] + spos[1] * W_emb[64 + t] + b_emb[t];
    __syncthreads();

    float gi[3], gh[3];
#pragma unroll
    for (int g = 0; g < 3; ++g) {
        const float* wi = W_ih + (g * 64 + t) * 64;
        const float* wh = W_hh + (g * 64 + t) * 64;
        float a = b_ih[g * 64 + t], b = b_hh[g * 64 + t];
#pragma unroll 8
        for (int k = 0; k < 64; ++k) {
            a += semb[k] * wi[k];
            b += sh[k] * wh[k];
        }
        gi[g] = a; gh[g] = b;
    }
    float r = sigmoidf_(gi[0] + gh[0]);
    float z = sigmoidf_(gi[1] + gh[1]);
    float n = tanhf(gi[2] + r * gh[2]);
    h[i * 64 + t] = (1.0f - z) * n + z * sh[t];
    if (t < 2) {
        last[i * 2 + t] = spos[t];
        out[i * 40 + (8 + step) * 2 + t] = spos[t];
    }
}

// ---------------------------------------------------------------------------
extern "C" void kernel_launch(void* const* d_in, const int* in_sizes, int n_in,
                              void* d_out, int out_size, void* d_ws, size_t ws_size,
                              hipStream_t stream) {
    const float* hidden = (const float*)d_in[0];   // (1,B,H)
    const float* gt     = (const float*)d_in[1];   // (B,OBS,2)
    const float* W_emb  = (const float*)d_in[2];
    const float* b_emb  = (const float*)d_in[3];
    const float* W_ih   = (const float*)d_in[4];
    const float* W_hh   = (const float*)d_in[5];
    const float* b_ih   = (const float*)d_in[6];
    const float* b_hh   = (const float*)d_in[7];
    const float* W_pos  = (const float*)d_in[8];
    const float* b_pos  = (const float*)d_in[9];
    const float* W_rel  = (const float*)d_in[10];
    const float* b_rel  = (const float*)d_in[11];
    const float* W_pool = (const float*)d_in[12];
    const float* b_pool = (const float*)d_in[13];
    float* out = (float*)d_out;

    // Workspace carve-up (float units)
    float* h      = (float*)d_ws;              // B*H f32
    float* last   = h + BATCH * H;             // B*2 f32
    float* u      = last + BATCH * 2;          // B*64 f32
    float* pooled = u + BATCH * 64;            // B*64 f32
    _Float16* wp16 = (_Float16*)(pooled + BATCH * 64);   // 128*64 f16
    _Float16* h16  = wp16 + (HP + H) * H;                // B*64 f16

    k_init<<<128, 256, 0, stream>>>(gt, W_pool, h, last, wp16, out);

    for (int s = 0; s < PRE; ++s) {
        const float* hsp = (s == 0) ? hidden : h;   // spool input h
        k_prep<<<128, 256, 0, stream>>>(last, hsp, W_rel, u, h16);
        k_spool<<<BATCH, 256, 0, stream>>>(u, h16, b_rel, b_pool, wp16, pooled);
        k_head<<<BATCH, 64, 0, stream>>>(pooled, W_pos, b_pos, W_emb, b_emb,
                                         W_ih, W_hh, b_ih, b_hh,
                                         h, last, out, s);
    }
}
